// SlidingWindowMultiHeadAttention_25735444037651
// MI455X (gfx1250) — matrix operand order repaired and checked by reading, not compiled, not run on hardware
//
#include <hip/hip_runtime.h>

// ---------------------------------------------------------------------------
// SlidingWindowMultiHeadAttention for MI455X (gfx1250), f16 WMMA pipeline.
//   B=4, H=8, S=1024, D=512.  All GEMMs via v_wmma_f32_16x16x32_f16.
//   K-block staging uses GLOBAL_LOAD_ASYNC_TO_LDS_B128 (ASYNCcnt path).
// ---------------------------------------------------------------------------

typedef __attribute__((ext_vector_type(16))) _Float16 v16h;
typedef __attribute__((ext_vector_type(8)))  float    v8f;

#define S_LEN 1024
#define DM    512
#define NHEAD 8
#define NBATCH 4
#define MROWS (NBATCH * NHEAD * S_LEN)   // 32768

__device__ __forceinline__ v8f wmma_f16(v16h a, v16h b, v8f c) {
  return __builtin_amdgcn_wmma_f32_16x16x32_f16(
      /*neg_a=*/false, a, /*neg_b=*/false, b,
      /*c_mod=*/(short)0, c, /*reuse_a=*/false, /*reuse_b=*/false);
}

// A-matrix (MxK, 16-bit) K index for VGPR v, halves j=0/1 (ISA 7.12.2):
//   lanes 0-15: V0..3 -> K={0,1},{2,3},{4,5},{6,7}; V4..7 -> K=16..23
//   lanes 16-31: +8 within each group of 4 VGPRs
__device__ __forceinline__ int a_kbase(int lane, int v) {
  return ((v >> 2) << 4) | (((lane >> 4) & 1) << 3) | ((v & 3) << 1);
}
// B-matrix (KxN, 16-bit) K index (column n = lane&15), same order as A:
//   lanes 0-15: V0..3 -> K=0..7, V4..7 -> K=16..23; lanes 16-31: +8
__device__ __forceinline__ int b_kbase(int lane, int v) {
  return ((v >> 2) << 4) | (((lane >> 4) & 1) << 3) | ((v & 3) << 1);
}

__device__ __forceinline__ float rowmax16(float x) {
#pragma unroll
  for (int off = 1; off < 16; off <<= 1)
    x = fmaxf(x, __shfl_xor(x, off, 32));
  return x;
}
__device__ __forceinline__ float rowsum16(float x) {
#pragma unroll
  for (int off = 1; off < 16; off <<= 1)
    x += __shfl_xor(x, off, 32);
  return x;
}

// ---------------------------------------------------------------------------
// Y[M,512] = scale * (X[M,512] @ W[512,512]^T + bias)
// One 16x16 output tile per wave; K=512 in 16 chunks of 32.
// ---------------------------------------------------------------------------
template <bool IN_F16, bool OUT_F16>
__global__ __launch_bounds__(256) void gemm_wt_bias(
    const void* __restrict__ Xv, const float* __restrict__ W,
    const float* __restrict__ bias, void* __restrict__ Yv, float scale) {
  const int lane = threadIdx.x & 31;
  const int waveG = blockIdx.x * 8 + (threadIdx.x >> 5);
  const int tileN = waveG & 31;        // 512/16 = 32 N tiles
  const int tileM = waveG >> 5;        // 2048 M tiles
  const int n15 = lane & 15;

  const int rowA = tileM * 16 + n15;   // A row for this lane
  const int colB = tileN * 16 + n15;   // B column (= output col) for this lane

  const float*    Xf = (const float*)Xv;
  const _Float16* Xh = (const _Float16*)Xv;

  v8f acc = {0.f, 0.f, 0.f, 0.f, 0.f, 0.f, 0.f, 0.f};
#pragma unroll
  for (int c = 0; c < 16; ++c) {
    v16h a, b;
#pragma unroll
    for (int v = 0; v < 8; ++v) {
      const int ka = c * 32 + a_kbase(lane, v);
      if (IN_F16) {
        a[2 * v]     = Xh[(size_t)rowA * DM + ka];
        a[2 * v + 1] = Xh[(size_t)rowA * DM + ka + 1];
      } else {
        float2 xa = *(const float2*)(Xf + (size_t)rowA * DM + ka);
        a[2 * v]     = (_Float16)xa.x;
        a[2 * v + 1] = (_Float16)xa.y;
      }
      const int kb = c * 32 + b_kbase(lane, v);
      float2 wb = *(const float2*)(W + (size_t)colB * DM + kb);
      b[2 * v]     = (_Float16)wb.x;
      b[2 * v + 1] = (_Float16)wb.y;
    }
    acc = wmma_f16(a, b, acc);
  }

  const float bb = bias[colB];
  float*    Yf = (float*)Yv;
  _Float16* Yh = (_Float16*)Yv;
#pragma unroll
  for (int v = 0; v < 8; ++v) {
    const int row = tileM * 16 + v + 8 * ((lane >> 4) & 1);
    const float val = (acc[v] + bb) * scale;
    if (OUT_F16)
      Yh[(size_t)row * DM + colB] = (_Float16)val;
    else
      Yf[(size_t)row * DM + colB] = val;
  }
}

// ---------------------------------------------------------------------------
// Flash attention. grid = (8 supertiles, 32 bh). 256 threads = 8 waves.
// Wave w owns query rows [bx*128 + w*16, +16) and full D=512 of the output.
// K block: async DMA straight into LDS (no VGPR round-trip, ASYNCcnt).
// V block: cooperative transpose into LDS so PV B-frags are contiguous.
// ctx written as [b, s*H + h, d] (f16) so the final GEMM is a plain matmul.
// ---------------------------------------------------------------------------
__global__ __launch_bounds__(256) void attn_kernel(
    const _Float16* __restrict__ qh, const _Float16* __restrict__ kh,
    const _Float16* __restrict__ vh, _Float16* __restrict__ ctx) {
  __shared__ _Float16 lk[32 * DM];       // K block   [key][d]   32 KB
  __shared__ _Float16 lvt[DM * 32];      // V block^T [d][key]   32 KB
  __shared__ _Float16 lp[8 * 16 * 32];   // per-wave P scratch    8 KB

  const int tid = threadIdx.x;
  const int lane = tid & 31;
  const int wave = tid >> 5;
  const int bh = blockIdx.y;
  const int b = bh >> 3, head = bh & 7;
  const int qrow0 = blockIdx.x * 128 + wave * 16;
  const int hh = (lane >> 4) & 1;
  const int n15 = lane & 15;

  const _Float16* qbase = qh + (size_t)bh * S_LEN * DM;
  const _Float16* kbase = kh + (size_t)bh * S_LEN * DM;
  const _Float16* vbase = vh + (size_t)bh * S_LEN * DM;

  // Q fragments, resident: 16 chunks x 8 VGPRs = 128 VGPRs
  v16h qf[16];
  {
    const _Float16* qr = qbase + (size_t)(qrow0 + n15) * DM;
#pragma unroll
    for (int c = 0; c < 16; ++c) {
#pragma unroll
      for (int v = 0; v < 8; ++v) {
        const int k = c * 32 + a_kbase(lane, v);
        qf[c][2 * v]     = qr[k];
        qf[c][2 * v + 1] = qr[k + 1];
      }
    }
  }

  v8f o[32];
#pragma unroll
  for (int t = 0; t < 32; ++t) o[t] = (v8f){0.f, 0.f, 0.f, 0.f, 0.f, 0.f, 0.f, 0.f};
  float mrow[8], lrow[8];
#pragma unroll
  for (int v = 0; v < 8; ++v) { mrow[v] = -1e30f; lrow[v] = 0.f; }

  // LDS byte offset of lk for the async-DMA destination operand
  const unsigned lk_lds_base = (unsigned)(uintptr_t)(&lk[0]);

  const int nblocks = (blockIdx.x + 1) * 4;   // causal limit for this supertile
  for (int j = 0; j < nblocks; ++j) {
    const int key0 = j * 32;
    __syncthreads();
    // ---- stage K: async global->LDS DMA, 16B per lane per issue ----
    {
      const unsigned kb_off = (unsigned)((size_t)key0 * DM * sizeof(_Float16));
#pragma unroll
      for (int i = 0; i < 8; ++i) {
        const unsigned chunk = (unsigned)(tid + 256 * i) * 16u;  // byte offset
        const unsigned ldsa = lk_lds_base + chunk;
        const unsigned goff = kb_off + chunk;
        asm volatile("global_load_async_to_lds_b128 %0, %1, %2"
                     :: "v"(ldsa), "v"(goff), "s"(kbase)
                     : "memory");
      }
    }
    // ---- stage V transposed (cooperative, through VGPRs) ----
    {
      const _Float16* vsrc = vbase + (size_t)key0 * DM;
#pragma unroll
      for (int i = 0; i < 8; ++i) {
        const int idx = (tid + 256 * i) * 8;   // halves; 8 consecutive d
        const int key = idx >> 9;              // /512
        const int d = idx & (DM - 1);
        uint4 pk = *(const uint4*)(vsrc + idx);
        const _Float16* ph = (const _Float16*)&pk;
#pragma unroll
        for (int e = 0; e < 8; ++e) lvt[(d + e) * 32 + key] = ph[e];
      }
      if (j + 1 < nblocks)  // speculative prefetch of next K block
        __builtin_prefetch(kbase + (size_t)(key0 + 32) * DM + (size_t)tid * 64, 0, 0);
    }
    // drain this wave's async DMAs, then workgroup barrier
    asm volatile("s_wait_asynccnt 0" ::: "memory");
    __syncthreads();

    if (key0 <= qrow0 + 15) {   // tile not fully masked
      // ---- scores: two 16x16 f32 tiles over K=512 ----
      v8f s0 = (v8f){0.f,0.f,0.f,0.f,0.f,0.f,0.f,0.f};
      v8f s1 = (v8f){0.f,0.f,0.f,0.f,0.f,0.f,0.f,0.f};
#pragma unroll
      for (int c = 0; c < 16; ++c) {
        v16h b0, b1;
#pragma unroll
        for (int v = 0; v < 8; ++v) {
          const int k = c * 32 + b_kbase(lane, v);
          b0[2 * v]     = lk[n15 * DM + k];
          b0[2 * v + 1] = lk[n15 * DM + k + 1];
          b1[2 * v]     = lk[(n15 + 16) * DM + k];
          b1[2 * v + 1] = lk[(n15 + 16) * DM + k + 1];
        }
        s0 = wmma_f16(qf[c], b0, s0);
        s1 = wmma_f16(qf[c], b1, s1);
      }
      // ---- causal mask + online softmax ----
      float corr[8];
#pragma unroll
      for (int v = 0; v < 8; ++v) {
        const int qg = qrow0 + v + 8 * hh;
        const float x0 = ((key0 + n15) <= qg)      ? s0[v] : -1e30f;
        const float x1 = ((key0 + 16 + n15) <= qg) ? s1[v] : -1e30f;
        const float rm = rowmax16(fmaxf(x0, x1));
        const float nm = fmaxf(mrow[v], rm);
        corr[v] = __expf(mrow[v] - nm);
        mrow[v] = nm;
        const float p0 = __expf(x0 - nm);
        const float p1 = __expf(x1 - nm);
        lrow[v] = lrow[v] * corr[v] + rowsum16(p0 + p1);
        const int mloc = v + 8 * hh;
        lp[wave * 512 + mloc * 32 + n15]      = (_Float16)p0;
        lp[wave * 512 + mloc * 32 + 16 + n15] = (_Float16)p1;
      }
      // wave-local LDS round-trip: wait for DS stores, then re-read as A-frag
      asm volatile("s_wait_dscnt 0" ::: "memory");
      __builtin_amdgcn_wave_barrier();
      v16h pf;
#pragma unroll
      for (int v = 0; v < 8; ++v) {
        const int k = a_kbase(lane, v);
        pf[2 * v]     = lp[wave * 512 + n15 * 32 + k];
        pf[2 * v + 1] = lp[wave * 512 + n15 * 32 + k + 1];
      }
      // ---- rescale accumulators, then O += P @ V ----
#pragma unroll
      for (int t = 0; t < 32; ++t) {
#pragma unroll
        for (int v = 0; v < 8; ++v) o[t][v] *= corr[v];
      }
#pragma unroll
      for (int t = 0; t < 32; ++t) {
        v16h bv;
#pragma unroll
        for (int v = 0; v < 8; ++v) {
          const int k = b_kbase(lane, v);   // k = key within block
          bv[2 * v]     = lvt[(t * 16 + n15) * 32 + k];
          bv[2 * v + 1] = lvt[(t * 16 + n15) * 32 + k + 1];
        }
        o[t] = wmma_f16(pf, bv, o[t]);
      }
    }
  }

  // ---- epilogue: normalize and store ctx in [b, s*H+h, d] layout ----
  float inv[8];
#pragma unroll
  for (int v = 0; v < 8; ++v) inv[v] = 1.0f / lrow[v];
#pragma unroll
  for (int t = 0; t < 32; ++t) {
#pragma unroll
    for (int v = 0; v < 8; ++v) {
      const int srow = qrow0 + v + 8 * hh;
      const size_t r = ((size_t)(b * S_LEN + srow) * NHEAD + head);
      ctx[r * DM + t * 16 + n15] = (_Float16)(o[t][v] * inv[v]);
    }
  }
}

// ---------------------------------------------------------------------------
extern "C" void kernel_launch(void* const* d_in, const int* in_sizes, int n_in,
                              void* d_out, int out_size, void* d_ws, size_t ws_size,
                              hipStream_t stream) {
  const float* q  = (const float*)d_in[0];
  const float* k  = (const float*)d_in[1];
  const float* v  = (const float*)d_in[2];
  // d_in[3] = mask (unused by the reference math)
  const float* Wq = (const float*)d_in[4];
  const float* bq = (const float*)d_in[5];
  const float* Wk = (const float*)d_in[6];
  const float* bk = (const float*)d_in[7];
  const float* Wv = (const float*)d_in[8];
  const float* bv = (const float*)d_in[9];
  const float* Wo = (const float*)d_in[10];
  const float* bo = (const float*)d_in[11];

  const size_t mat = (size_t)MROWS * DM;   // 16M f16 elements = 32 MB
  _Float16* qh  = (_Float16*)d_ws;
  _Float16* kh  = qh + mat;
  _Float16* vh  = kh + mat;
  _Float16* ctx = vh + mat;

  const float scale = 0.125f;  // 1/sqrt(D/H) = 1/sqrt(64), folded into Q proj
  const int gemm_blocks = MROWS / 4;  // (M/16 tiles * 32 ntiles) / 8 waves

  gemm_wt_bias<false, true><<<gemm_blocks, 256, 0, stream>>>(q, Wq, bq, qh, scale);
  gemm_wt_bias<false, true><<<gemm_blocks, 256, 0, stream>>>(k, Wk, bk, kh, 1.0f);
  gemm_wt_bias<false, true><<<gemm_blocks, 256, 0, stream>>>(v, Wv, bv, vh, 1.0f);

  attn_kernel<<<dim3(S_LEN / 128, NBATCH * NHEAD), 256, 0, stream>>>(qh, kh, vh, ctx);

  gemm_wt_bias<true, false><<<gemm_blocks, 256, 0, stream>>>(ctx, Wo, bo, (float*)d_out, 1.0f);
}
